// Linear_34428457845432
// MI455X (gfx1250) — matrix operand order repaired and checked by reading, not compiled, not run on hardware
//
#include <hip/hip_runtime.h>
#include <stdint.h>

typedef __attribute__((ext_vector_type(16))) _Float16 v16h;
typedef __attribute__((ext_vector_type(8)))  _Float16 v8h;
typedef __attribute__((ext_vector_type(8)))  float    v8f;
typedef __attribute__((ext_vector_type(4)))  unsigned int u32x4;
typedef __attribute__((ext_vector_type(4)))  int      i32x4;
typedef __attribute__((ext_vector_type(8)))  int      i32x8;

#define B_DIM   512
#define IN_DIM  4096
#define OUT_DIM 4096
#define QMAXF   127.0f

#define BM 32            // C-tile rows per workgroup (batch dim)
#define BN 128           // C-tile cols per workgroup (out dim)
#define BK 64            // K chunk staged in LDS
#define LDS_STRIDE 72    // halves; 64 + 8 pad (matches TDM pad fields below)

// ---------------------------------------------------------------------------
// Kernel 1: per-sample symmetric 8-bit quantize of x.
// Stores integer levels q in [-127,127] exactly as f16, and scale[b] as f32.
// out[b,o] = scale[b] * sum_k W[o,k]*q[b,k]  (scale applied in GEMM epilogue)
// ---------------------------------------------------------------------------
__global__ __launch_bounds__(256)
void quantize_x_kernel(const float* __restrict__ x,
                       _Float16* __restrict__ xh,
                       float* __restrict__ scales) {
  __shared__ float red[256];
  const int b = blockIdx.x;
  const int t = threadIdx.x;
  const float* xb = x + (size_t)b * IN_DIM;

  float4 v[4];
  float m = 0.0f;
#pragma unroll
  for (int i = 0; i < 4; ++i) {
    v[i] = ((const float4*)xb)[i * 256 + t];
    m = fmaxf(m, fmaxf(fmaxf(fabsf(v[i].x), fabsf(v[i].y)),
                       fmaxf(fabsf(v[i].z), fabsf(v[i].w))));
  }
  red[t] = m;
  __syncthreads();
#pragma unroll
  for (int s = 128; s > 0; s >>= 1) {
    if (t < s) red[t] = fmaxf(red[t], red[t + s]);
    __syncthreads();
  }
  const float scale = fmaxf(red[0] / QMAXF, 1e-12f);
  if (t == 0) scales[b] = scale;
  const float inv = 1.0f / scale;

#pragma unroll
  for (int i = 0; i < 4; ++i) {
    const int e = (i * 256 + t) * 4;
    float q0 = fminf(fmaxf(rintf(v[i].x * inv), -QMAXF), QMAXF);
    float q1 = fminf(fmaxf(rintf(v[i].y * inv), -QMAXF), QMAXF);
    float q2 = fminf(fmaxf(rintf(v[i].z * inv), -QMAXF), QMAXF);
    float q3 = fminf(fmaxf(rintf(v[i].w * inv), -QMAXF), QMAXF);
    _Float16* p = xh + (size_t)b * IN_DIM + e;
    p[0] = (_Float16)q0; p[1] = (_Float16)q1;
    p[2] = (_Float16)q2; p[3] = (_Float16)q3;
  }
}

// ---------------------------------------------------------------------------
// Kernel 2: W f32 -> f16 (|W| ~ 2^-12, comfortably in f16 normal range).
// ---------------------------------------------------------------------------
__global__ __launch_bounds__(256)
void convert_w_kernel(const float* __restrict__ W, _Float16* __restrict__ Wh) {
  const size_t idx = ((size_t)blockIdx.x * 256 + threadIdx.x) * 8;
  float4 a = ((const float4*)(W + idx))[0];
  float4 c = ((const float4*)(W + idx))[1];
  v8h h;
  h[0] = (_Float16)a.x; h[1] = (_Float16)a.y;
  h[2] = (_Float16)a.z; h[3] = (_Float16)a.w;
  h[4] = (_Float16)c.x; h[5] = (_Float16)c.y;
  h[6] = (_Float16)c.z; h[7] = (_Float16)c.w;
  *(v8h*)(Wh + idx) = h;
}

// ---------------------------------------------------------------------------
// Async global -> LDS, 16B per lane (ASYNCcnt path) -- used for the A tile.
// ---------------------------------------------------------------------------
__device__ __forceinline__ void async_copy_b128(uint32_t lds_addr,
                                                const void* gaddr) {
  asm volatile("global_load_async_to_lds_b128 %0, %1, off"
               :: "v"(lds_addr), "v"(gaddr)
               : "memory");
}

__device__ __forceinline__ uint32_t lds_lo32(const void* p) {
  return (uint32_t)(uintptr_t)p;   // generic LDS addr: low 32 bits give the LDS address
}

// ---------------------------------------------------------------------------
// TDM: one descriptor DMAs a rows x 64-half 2D tile (row stride IN_DIM
// halves) from global to LDS, inserting 4 DWORDs of padding after every
// 32 DWORDs (one row) => LDS row stride = 72 halves = LDS_STRIDE.
//   D# group0: count=1 | lds_addr | global_addr | type=2
//   D# group1: data_size=2B, pad_enable, pad_interval=code4(32 DW),
//              pad_amount=code3(4 DW), tensor_dim0=IN, tensor_dim1=OUT,
//              tile_dim0=64, tile_dim1=rows, dim0_stride=IN
// ---------------------------------------------------------------------------
__device__ __forceinline__ void tdm_load_tile(uint32_t lds_addr,
                                              uint64_t gaddr,
                                              int rows) {
  u32x4 g0;
  g0[0] = 1u;                                     // count=1, user descriptor
  g0[1] = lds_addr;                               // lds_addr (bytes)
  g0[2] = (unsigned int)gaddr;                    // global_addr[31:0]
  g0[3] = (unsigned int)(gaddr >> 32) | (2u << 30); // addr[56:32] | type=2

  i32x8 g1;
  g1[0] = (1 << 16)        // data_size = 2 bytes
        | (1 << 20)        // pad_enable
        | (4 << 22)        // pad_interval code 4 -> every 32 DWORDs
        | (3 << 25);       // pad_amount  code 3 -> 4 DWORDs
  g1[1] = (int)((unsigned)IN_DIM << 16);   // tensor_dim0[15:0] in [63:48]
  g1[2] = (int)((unsigned)OUT_DIM << 16);  // dim0 hi=0 | tensor_dim1[15:0]
  g1[3] = (64 << 16);                      // dim1 hi=0 | tile_dim0=64
  g1[4] = rows;                            // tile_dim1=rows | tile_dim2=0
  g1[5] = IN_DIM;                          // tensor_dim0_stride lo32
  g1[6] = 0;                               // stride hi | dim1_stride lo
  g1[7] = 0;

  i32x4 z4 = {0, 0, 0, 0};
#if __clang_major__ >= 23
  i32x8 z8 = {0, 0, 0, 0, 0, 0, 0, 0};
  __builtin_amdgcn_tensor_load_to_lds(g0, g1, z4, z4, z8, 0);
#else
  __builtin_amdgcn_tensor_load_to_lds(g0, g1, z4, z4, 0);
#endif
}

// ---------------------------------------------------------------------------
// Kernel 3: WMMA GEMM.  C[M=512, N=4096] = Q[512,4096] x W^T, K=4096.
// 8 waves / workgroup; tile 32(M) x 128(N); each wave owns 16x32 of C
// (4 WMMAs per K-chunk).  Double-buffered LDS:
//   A tile (4KB/chunk): per-thread global_load_async_to_lds_b128 (ASYNCcnt)
//   B tile (16KB/chunk): single TENSOR_LOAD_TO_LDS descriptor from wave 0
//                        (TENSORcnt), freeing the VMEM pipe during compute.
// Fragment k order: contiguous k = 16*(lane/16) + s, same for A and B (wave32).
// ---------------------------------------------------------------------------
__global__ __launch_bounds__(256)
void gemm_wmma_kernel(const _Float16* __restrict__ Xh,
                      const _Float16* __restrict__ Wh,
                      const float* __restrict__ scales,
                      float* __restrict__ out) {
  __shared__ _Float16 As[2][BM * LDS_STRIDE];
  __shared__ _Float16 Bs[2][BN * LDS_STRIDE];

  const int t    = threadIdx.x;
  const int wave = t >> 5;
  const int lane = t & 31;
  const int m0   = blockIdx.y * BM;
  const int n0   = blockIdx.x * BN;

  const int mi  = wave >> 2;          // 0..1  (M sub-tile)
  const int nw  = wave & 3;           // 0..3  (N wave column)
  const int l15 = lane & 15;
  const int hi  = (lane >> 4) & 1;

  const int a_row  = 16 * mi + l15;
  const int b_row0 = 32 * nw + l15;
  const int a_koff = hi ? 16 : 0;
  const int b_koff = hi ? 16 : 0;

  // A staging: 32x64 halves = 4KB -> one b128 per thread
  const int ar = t >> 3;              // 0..31
  const int ac = (t & 7) * 8;         // 0..56
  const _Float16* gA = &Xh[(size_t)(m0 + ar) * IN_DIM + ac];

  uint32_t ldsA[2], ldsBb[2];
#pragma unroll
  for (int b = 0; b < 2; ++b) {
    ldsA[b]  = lds_lo32(&As[b][ar * LDS_STRIDE + ac]);
    ldsBb[b] = lds_lo32(&Bs[b][0]);
  }
  const uint64_t gB_base = (uint64_t)(uintptr_t)&Wh[(size_t)n0 * IN_DIM];

  // wave-uniform scalar predicate -> scalar branch (TDM ignores EXEC)
  const bool is_w0 = (__builtin_amdgcn_readfirstlane(t) < 32);

  v8f acc0 = {};
  v8f acc1 = {};

  // prologue: chunk 0 in flight
  async_copy_b128(ldsA[0], gA);
  if (is_w0) tdm_load_tile(ldsBb[0], gB_base, BN);

  for (int k0 = 0; k0 < IN_DIM; k0 += BK) {
    const int  buf      = (k0 >> 6) & 1;
    const bool has_next = (k0 + BK) < IN_DIM;

    if (has_next) {
      async_copy_b128(ldsA[buf ^ 1], gA + k0 + BK);
      if (is_w0) tdm_load_tile(ldsBb[buf ^ 1],
                               gB_base + (uint64_t)(k0 + BK) * 2, BN);
      asm volatile("s_wait_asynccnt 0x1" ::: "memory");  // A chunk k0 done
      __builtin_amdgcn_s_wait_tensorcnt(1);              // B chunk k0 done
    } else {
      asm volatile("s_wait_asynccnt 0x0" ::: "memory");
      __builtin_amdgcn_s_wait_tensorcnt(0);
    }
    __syncthreads();   // chunk k0 resident in LDS[buf] for all waves

#pragma unroll
    for (int kk = 0; kk < BK; kk += 32) {
      v8h alo = *(const v8h*)(&As[buf][a_row * LDS_STRIDE + kk + a_koff]);
      v8h ahi = *(const v8h*)(&As[buf][a_row * LDS_STRIDE + kk + a_koff + 8]);
      v16h a;
#pragma unroll
      for (int i = 0; i < 8; ++i) { a[i] = alo[i]; a[i + 8] = ahi[i]; }

#pragma unroll
      for (int sub = 0; sub < 2; ++sub) {
        const int b_row = b_row0 + 16 * sub;
        v8h blo = *(const v8h*)(&Bs[buf][b_row * LDS_STRIDE + kk + b_koff]);
        v8h bhi = *(const v8h*)(&Bs[buf][b_row * LDS_STRIDE + kk + b_koff + 8]);
        v16h bb;
#pragma unroll
        for (int i = 0; i < 8; ++i) { bb[i] = blo[i]; bb[i + 8] = bhi[i]; }
        if (sub == 0) {
          acc0 = __builtin_amdgcn_wmma_f32_16x16x32_f16(
              false, a, false, bb, (short)0, acc0, false, false);
        } else {
          acc1 = __builtin_amdgcn_wmma_f32_16x16x32_f16(
              false, a, false, bb, (short)0, acc1, false, false);
        }
      }
    }
    __syncthreads();   // LDS[buf] free for chunk k0 + 2*BK
  }

  // ---- epilogue: per-sample dequant scale, f32 store ----
#pragma unroll
  for (int sub = 0; sub < 2; ++sub) {
    const int n = n0 + 32 * nw + 16 * sub + l15;
    const v8f& acc = sub ? acc1 : acc0;
#pragma unroll
    for (int r = 0; r < 8; ++r) {
      const int m = m0 + 16 * mi + r + hi * 8;
      out[(size_t)m * OUT_DIM + n] = acc[r] * scales[m];
    }
  }
}

// ---------------------------------------------------------------------------
// Workspace layout (d_ws):
//   [0, 32MB)        Wh : f16[OUT*IN]
//   [32MB, 36MB)     Xh : f16[B*IN]   (quantized integer levels)
//   [36MB, +2KB)     scales : f32[B]
// ---------------------------------------------------------------------------
extern "C" void kernel_launch(void* const* d_in, const int* in_sizes, int n_in,
                              void* d_out, int out_size, void* d_ws, size_t ws_size,
                              hipStream_t stream) {
  const float* x = (const float*)d_in[0];   // [B, IN, 1] f32
  const float* W = (const float*)d_in[1];   // [OUT, IN]  f32
  float* out = (float*)d_out;               // [B, OUT, 1] f32

  char* ws = (char*)d_ws;
  _Float16* Wh = (_Float16*)ws;
  _Float16* Xh = (_Float16*)(ws + (size_t)OUT_DIM * IN_DIM * sizeof(_Float16));
  float* scales = (float*)(ws + (size_t)OUT_DIM * IN_DIM * sizeof(_Float16)
                              + (size_t)B_DIM * IN_DIM * sizeof(_Float16));

  quantize_x_kernel<<<B_DIM, 256, 0, stream>>>(x, Xh, scales);

  const int wblocks = (int)(((size_t)OUT_DIM * IN_DIM) / (256 * 8)); // 8192
  convert_w_kernel<<<wblocks, 256, 0, stream>>>(W, Wh);

  dim3 grid(OUT_DIM / BN, B_DIM / BM);  // (32, 16)
  gemm_wmma_kernel<<<grid, 256, 0, stream>>>(Xh, Wh, scales, out);
}
